// FRU_48653389529467
// MI455X (gfx1250) — matrix operand order repaired and checked by reading, not compiled, not run on hardware
//
#include <hip/hip_runtime.h>
#include <math.h>

// ---------------------------------------------------------------------------
// FRU forward for MI455X (gfx1250, wave32, WMMA).
//   proj = x @ w_in^T              (bf16x3 WMMA GEMM, LDS bf16 hi/lo planes,
//                                   register double-buffered global loads)
//   prep: k-normalize, forget-gate sigmoid (in-place on proj)
//   scan: per-head recurrence, state@W on WMMA, state resident in LDS,
//         one barrier per step (ping-pong y reduction buffer)
//   gate/silu + RMS norm
//   out  = yn @ w_out^T            (same GEMM)
// ---------------------------------------------------------------------------

typedef __attribute__((ext_vector_type(16))) __bf16 v16bf;
typedef __attribute__((ext_vector_type(8)))  __bf16 v8bf;
typedef __attribute__((ext_vector_type(8)))  float  v8f;

#define S_LEN    1024
#define D_IN     2048
#define D_OUT    2048
#define STATE_D  1024
#define NH       16
#define DH       64
#define CONV     4096
#define PROJ_D   5120   // CONV + STATE_D

// ---- ISA fragment layouts (CDNA5 ISA §7.12.2) -------------------
// A (16x32, 16-bit): lane&15 = row; element e -> VGPR v=e/2, slot=e&1:
//   K = (v<4 ? 2v : 2v+8) + 8*(lane>>4) + slot
//   => per lane two contiguous K runs: [8h, 8h+8) and [16+8h, 16+8h+8)
__device__ __forceinline__ int a_frag_k(int e, int half) {
  int v = e >> 1, slot = e & 1;
  int kb = (v < 4) ? (2 * v) : (2 * v + 8);
  return kb + 8 * half + slot;
}
// B (32x16, 16-bit): lane&15 = column; element e -> K = a_frag_k(e, lane>>4), as for A
//   => per lane two contiguous K runs: [8h, 8h+8) and [16+8h, 16+8h+8)
// C/D (16x16 f32): col = lane&15, VGPR v -> row = v + 8*(lane>>4).

__device__ __forceinline__ void split_bf16(float f, __bf16& hi, __bf16& lo) {
  hi = (__bf16)f;
  lo = (__bf16)(f - (float)hi);
}

// Two contiguous 8x-bf16 LDS reads -> one 16-elem fragment (zero VALU).
__device__ __forceinline__ v16bf ld_frag(const __bf16* row, int off0, int off1) {
  v8bf a = *(const v8bf*)(row + off0);
  v8bf b = *(const v8bf*)(row + off1);
  return __builtin_shufflevector(a, b, 0, 1, 2, 3, 4, 5, 6, 7,
                                       8, 9, 10, 11, 12, 13, 14, 15);
}

#define WMMA_BF16(A, B, C) \
  __builtin_amdgcn_wmma_f32_16x16x32_bf16(false, (A), false, (B), (short)0, (C), false, false)

// ---------------------------------------------------------------------------
// GEMM (NT): C[M,N] = A[M,K] * B[N,K]^T, K-contiguous operands, f32 in/out.
// bf16x3 fp32-emulation on the matrix cores. 128x128 tile, 8 waves, each wave
// a 16x128 strip. f32 -> bf16 hi/lo split happens ONCE at staging; fragments
// are pure ds_load_b128. Next K-chunk's global loads are issued into
// registers before the WMMA block so their latency hides under the math.
// ---------------------------------------------------------------------------
#define BM 128
#define BN 128
#define BK 32
#define LDK (BK + 8)   // bf16 stride: 40*2B = 80B, multiple of 16B

__global__ __launch_bounds__(256)
void fru_gemm_nt(const float* __restrict__ A, const float* __restrict__ B,
                 float* __restrict__ C, int M, int N, int K) {
  __shared__ __bf16 Ah[BM][LDK], Al[BM][LDK];
  __shared__ __bf16 Bh[BN][LDK], Bl[BN][LDK];

  const int tid  = threadIdx.x;
  const int wave = tid >> 5;
  const int lane = tid & 31;
  const int half = lane >> 4;
  const int ml   = lane & 15;
  const int mblk = blockIdx.y * BM;
  const int nblk = blockIdx.x * BN;

  // Per-thread staging slots: j = tid + i*256 -> row = j>>3, kk = (j&7)*4.
  const int srow = tid >> 3;          // +32 per i
  const int skk  = (tid & 7) * 4;

  v8f acc[8];
#pragma unroll
  for (int i = 0; i < 8; ++i) acc[i] = (v8f){};

  // Preload chunk 0 into registers.
  float4 ra[4], rb[4];
#pragma unroll
  for (int i = 0; i < 4; ++i) {
    ra[i] = *(const float4*)(A + (size_t)(mblk + srow + 32 * i) * K + skk);
    rb[i] = *(const float4*)(B + (size_t)(nblk + srow + 32 * i) * K + skk);
  }

  for (int k0 = 0; k0 < K; k0 += BK) {
    // Split staged registers into LDS bf16 hi/lo planes.
#pragma unroll
    for (int i = 0; i < 4; ++i) {
      int row = srow + 32 * i;
      union { __bf16 b[4]; uint2 u; } ph, pl;
      split_bf16(ra[i].x, ph.b[0], pl.b[0]);
      split_bf16(ra[i].y, ph.b[1], pl.b[1]);
      split_bf16(ra[i].z, ph.b[2], pl.b[2]);
      split_bf16(ra[i].w, ph.b[3], pl.b[3]);
      *(uint2*)&Ah[row][skk] = ph.u;
      *(uint2*)&Al[row][skk] = pl.u;
      split_bf16(rb[i].x, ph.b[0], pl.b[0]);
      split_bf16(rb[i].y, ph.b[1], pl.b[1]);
      split_bf16(rb[i].z, ph.b[2], pl.b[2]);
      split_bf16(rb[i].w, ph.b[3], pl.b[3]);
      *(uint2*)&Bh[row][skk] = ph.u;
      *(uint2*)&Bl[row][skk] = pl.u;
    }
    __syncthreads();

    // Issue next chunk's global loads now; waits land at next LDS store.
    if (k0 + BK < K) {
#pragma unroll
      for (int i = 0; i < 4; ++i) {
        ra[i] = *(const float4*)(A + (size_t)(mblk + srow + 32 * i) * K + k0 + BK + skk);
        rb[i] = *(const float4*)(B + (size_t)(nblk + srow + 32 * i) * K + k0 + BK + skk);
      }
    }

    // Fragments: pure b128 LDS reads.
    v16bf ahi = ld_frag(&Ah[16 * wave + ml][0], 8 * half, 16 + 8 * half);
    v16bf alo = ld_frag(&Al[16 * wave + ml][0], 8 * half, 16 + 8 * half);
    v16bf bhi[8], blo[8];
#pragma unroll
    for (int nt = 0; nt < 8; ++nt) {
      bhi[nt] = ld_frag(&Bh[nt * 16 + ml][0], 8 * half, 16 + 8 * half);
      blo[nt] = ld_frag(&Bl[nt * 16 + ml][0], 8 * half, 16 + 8 * half);
    }
    // Term-outer, tile-inner: consecutive WMMAs hit independent accumulators.
#pragma unroll
    for (int nt = 0; nt < 8; ++nt) acc[nt] = WMMA_BF16(ahi, bhi[nt], acc[nt]);
#pragma unroll
    for (int nt = 0; nt < 8; ++nt) acc[nt] = WMMA_BF16(alo, bhi[nt], acc[nt]);
#pragma unroll
    for (int nt = 0; nt < 8; ++nt) acc[nt] = WMMA_BF16(ahi, blo[nt], acc[nt]);
    __syncthreads();
  }

  // Write back: row = 16*wave + 8*half + v, col = nt*16 + ml.
#pragma unroll
  for (int nt = 0; nt < 8; ++nt) {
#pragma unroll
    for (int v = 0; v < 8; ++v) {
      int row = mblk + 16 * wave + 8 * half + v;
      int col = nblk + nt * 16 + ml;
      C[(size_t)row * N + col] = acc[nt][v];
    }
  }
}

// ---------------------------------------------------------------------------
// Prep: in-place on proj. Normalize k per (s,h); f -> sigmoid forget gate.
// ---------------------------------------------------------------------------
__global__ __launch_bounds__(32)
void fru_prep(float* __restrict__ proj, const float* __restrict__ fm,
              const float* __restrict__ fb) {
  const int s = blockIdx.x >> 4;
  const int h = blockIdx.x & 15;
  const int lane = threadIdx.x;
  float* p = proj + (size_t)s * PROJ_D;

  float* kp = p + STATE_D + h * DH;
  float k0 = kp[lane], k1 = kp[lane + 32];
  float ss = k0 * k0 + k1 * k1;
#pragma unroll
  for (int off = 16; off > 0; off >>= 1) ss += __shfl_xor(ss, off, 32);
  float inv = 1.0f / fmaxf(sqrtf(ss), 1e-12f);
  kp[lane] = k0 * inv;
  kp[lane + 32] = k1 * inv;

  float m2 = 2.0f / (1.0f + expf(-fm[h]));
  float* fp = p + 3 * STATE_D + h * DH;
  const float* fbh = fb + h * DH;
#pragma unroll
  for (int i = 0; i < 2; ++i) {
    int d = lane + i * 32;
    float fv = m2 * (fp[d] + fbh[d]);
    fp[d] = 1.0f / (1.0f + expf(-fv));
  }
}

// ---------------------------------------------------------------------------
// Scan: one block per head (4 waves). Wave r owns state rows [16r,16r+16):
// the row-block recurrence is wave-local (no cross-wave state traffic); only
// y = q^T state needs a block reduction — done through a ping-pong LDS
// buffer so each step needs exactly ONE barrier. W fragments (hi+lo,
// 2 K-chunks x 4 N-tiles) are register-resident for all 1024 steps. proj
// (20MB) is L2-resident (192MB) so per-step vector fetches are L2 hits.
// ---------------------------------------------------------------------------
__global__ __launch_bounds__(128)
void fru_scan(const float* __restrict__ proj, const float* __restrict__ sw,
              float* __restrict__ y) {
  __shared__ float st[64 * 68];        // state[row][col], stride 68
  __shared__ float ypart[2][4][64];    // [step parity][wave][col]

  const int h    = blockIdx.x;
  const int tid  = threadIdx.x;
  const int wave = tid >> 5;
  const int lane = tid & 31;
  const int half = lane >> 4;
  const int ml   = lane & 15;
  const int rowBase = 16 * wave;

  for (int i = tid; i < 64 * 68; i += 128) st[i] = 0.0f;

  // W fragments (B-operand layout), split hi/lo, kept in registers.
  const float* W = sw + (size_t)h * DH * DH;   // W[d][e] row-major
  v16bf wbhi[2][4], wblo[2][4];
#pragma unroll
  for (int kc = 0; kc < 2; ++kc) {
#pragma unroll
    for (int tj = 0; tj < 4; ++tj) {
#pragma unroll
      for (int e = 0; e < 16; ++e) {
        int d = 32 * kc + a_frag_k(e, half);
        float f = W[d * DH + tj * 16 + ml];
        __bf16 hi, lo; split_bf16(f, hi, lo);
        wbhi[kc][tj][e] = hi; wblo[kc][tj][e] = lo;
      }
    }
  }
  __syncthreads();

  const int rb = rowBase + 8 * half;   // first C-layout row this lane owns

  for (int t = 0; t < S_LEN; ++t) {
    const float* p = proj + (size_t)t * PROJ_D;
    float qv[8], knv[8], fgv[8], vv[4];
#pragma unroll
    for (int v = 0; v < 8; ++v) {
      int idx = h * DH + rb + v;
      qv[v]  = p[idx];
      knv[v] = p[STATE_D + idx];
      fgv[v] = p[3 * STATE_D + idx];
    }
#pragma unroll
    for (int tj = 0; tj < 4; ++tj) vv[tj] = p[2 * STATE_D + h * DH + tj * 16 + ml];

    // A fragments: this wave's 16-row strip of state, both K chunks.
    v16bf ahi[2], alo[2];
#pragma unroll
    for (int kc = 0; kc < 2; ++kc) {
#pragma unroll
      for (int e = 0; e < 16; ++e) {
        float f = st[(rowBase + ml) * 68 + 32 * kc + a_frag_k(e, half)];
        __bf16 hi, lo; split_bf16(f, hi, lo);
        ahi[kc][e] = hi; alo[kc][e] = lo;
      }
    }

    // state @ W: two accumulator sets (per K-chunk), term-outer ordering ->
    // 8 independent WMMAs between any dependent pair, chain depth 3.
    v8f a0[4], a1[4];
#pragma unroll
    for (int tj = 0; tj < 4; ++tj) a0[tj] = WMMA_BF16(ahi[0], wbhi[0][tj], (v8f){});
#pragma unroll
    for (int tj = 0; tj < 4; ++tj) a1[tj] = WMMA_BF16(ahi[1], wbhi[1][tj], (v8f){});
#pragma unroll
    for (int tj = 0; tj < 4; ++tj) a0[tj] = WMMA_BF16(alo[0], wbhi[0][tj], a0[tj]);
#pragma unroll
    for (int tj = 0; tj < 4; ++tj) a1[tj] = WMMA_BF16(alo[1], wbhi[1][tj], a1[tj]);
#pragma unroll
    for (int tj = 0; tj < 4; ++tj) a0[tj] = WMMA_BF16(ahi[0], wblo[0][tj], a0[tj]);
#pragma unroll
    for (int tj = 0; tj < 4; ++tj) a1[tj] = WMMA_BF16(ahi[1], wblo[1][tj], a1[tj]);

    // Gate + rank-1 update, state write-back, y partials.
    float yp[4] = {0.f, 0.f, 0.f, 0.f};
#pragma unroll
    for (int tj = 0; tj < 4; ++tj) {
      v8f acc = a0[tj] + a1[tj];
#pragma unroll
      for (int v = 0; v < 8; ++v) {
        float ns = fgv[v] * acc[v] + knv[v] * vv[tj];
        st[(rb + v) * 68 + tj * 16 + ml] = ns;
        yp[tj] += qv[v] * ns;
      }
    }
#pragma unroll
    for (int tj = 0; tj < 4; ++tj) yp[tj] += __shfl_xor(yp[tj], 16, 32);
    const int pb = t & 1;
    if (half == 0) {
#pragma unroll
      for (int tj = 0; tj < 4; ++tj) ypart[pb][wave][tj * 16 + ml] = yp[tj];
    }
    __syncthreads();   // single barrier: next step writes the OTHER buffer
    if (tid < 64) {
      y[(size_t)t * STATE_D + h * DH + tid] =
          ypart[pb][0][tid] + ypart[pb][1][tid] +
          ypart[pb][2][tid] + ypart[pb][3][tid];
    }
  }
}

// ---------------------------------------------------------------------------
// y * silu(gate), RMS-norm over STATE, scale by g_norm_weight. One block/row.
// ---------------------------------------------------------------------------
__global__ __launch_bounds__(256)
void fru_gate_norm(const float* __restrict__ y, const float* __restrict__ proj,
                   const float* __restrict__ gw, float* __restrict__ yn) {
  __shared__ float red[8];
  const int s = blockIdx.x;
  const int tid = threadIdx.x;
  const float* yr = y + (size_t)s * STATE_D;
  const float* g  = proj + (size_t)s * PROJ_D + CONV;

  float vals[4], ss = 0.0f;
#pragma unroll
  for (int i = 0; i < 4; ++i) {
    int idx = tid + i * 256;
    float gv = g[idx];
    float v  = yr[idx] * (gv / (1.0f + expf(-gv)));
    vals[i] = v;
    ss += v * v;
  }
#pragma unroll
  for (int off = 16; off > 0; off >>= 1) ss += __shfl_xor(ss, off, 32);
  if ((tid & 31) == 0) red[tid >> 5] = ss;
  __syncthreads();
  float tot = 0.0f;
#pragma unroll
  for (int i = 0; i < 8; ++i) tot += red[i];
  float rms = rsqrtf(tot * (1.0f / STATE_D) + 1e-6f);
#pragma unroll
  for (int i = 0; i < 4; ++i) {
    int idx = tid + i * 256;
    yn[(size_t)s * STATE_D + idx] = vals[i] * rms * gw[idx];
  }
}

// ---------------------------------------------------------------------------
extern "C" void kernel_launch(void* const* d_in, const int* in_sizes, int n_in,
                              void* d_out, int out_size, void* d_ws, size_t ws_size,
                              hipStream_t stream) {
  const float* x     = (const float*)d_in[0];
  const float* w_in  = (const float*)d_in[1];
  const float* sw    = (const float*)d_in[2];
  const float* fm    = (const float*)d_in[3];
  const float* fb    = (const float*)d_in[4];
  const float* gw    = (const float*)d_in[5];
  const float* w_out = (const float*)d_in[6];
  float* out = (float*)d_out;

  char* ws = (char*)d_ws;
  float* proj = (float*)ws;                                         // 1024*5120 f32
  float* yb   = (float*)(ws + (size_t)S_LEN * PROJ_D * 4);          // 1024*1024 f32
  float* yn   = (float*)(ws + (size_t)S_LEN * PROJ_D * 4
                            + (size_t)S_LEN * STATE_D * 4);         // 1024*1024 f32

  dim3 g1(PROJ_D / BN, S_LEN / BM);
  fru_gemm_nt<<<g1, 256, 0, stream>>>(x, w_in, proj, S_LEN, PROJ_D, D_IN);

  fru_prep<<<S_LEN * NH, 32, 0, stream>>>(proj, fm, fb);

  fru_scan<<<NH, 128, 0, stream>>>(proj, sw, yb);

  fru_gate_norm<<<S_LEN, 256, 0, stream>>>(yb, proj, gw, yn);

  dim3 g2(D_OUT / BN, S_LEN / BM);
  fru_gemm_nt<<<g2, 256, 0, stream>>>(yn, w_out, out, S_LEN, D_OUT, STATE_D);
}
